// MultiHeadAttention_28071906247355
// MI455X (gfx1250) — matrix operand order repaired and checked by reading, not compiled, not run on hardware
//
#include <hip/hip_runtime.h>

// ---------------------------------------------------------------------------
// Types for CDNA5 WMMA (wave32): bf16 A/B fragments (16 x bf16 = 8 VGPRs),
// f32 accumulator (8 x f32 = 8 VGPRs).
// ---------------------------------------------------------------------------
typedef __bf16 bf16;
typedef __attribute__((ext_vector_type(4)))  bf16  v4bf;
typedef __attribute__((ext_vector_type(8)))  bf16  v8bf;
typedef __attribute__((ext_vector_type(16))) bf16  v16bf;
typedef __attribute__((ext_vector_type(8)))  float v8f;
typedef __attribute__((ext_vector_type(4)))  int   v4i;

union BF16x16 { v16bf v; v8bf h[2]; };

__device__ __forceinline__ v8f wmma_bf16(v16bf a, v16bf b, v8f c) {
  // D(16x16,f32) = A(16x32,bf16) * B(32x16,bf16) + C
  return __builtin_amdgcn_wmma_f32_16x16x32_bf16(
      /*neg_a=*/false, a, /*neg_b=*/false, b,
      /*c_mod=*/(short)0, c, /*reuse_a=*/false, /*reuse_b=*/false);
}

// A fragment (16x32, row-major source, leading dim `ld` in elements):
//   lane m = lane&15 ; half sel kb = (lane>=16)?8:0
//   elems 0..7  -> k = kb + 0..7 ; elems 8..15 -> k = kb + 16..23
__device__ __forceinline__ v16bf ld_frag_a(const bf16* tile, int ld, int lane) {
  const int m  = lane & 15;
  const int kb = (lane & 16) ? 8 : 0;
  const bf16* p = tile + m * ld + kb;
  BF16x16 r;
  r.h[0] = *(const v8bf*)(p);
  r.h[1] = *(const v8bf*)(p + 16);
  return r.v;
}

// B fragment (32x16) loaded from an N-major ("B transposed") tile [n][k]:
//   lane n = lane&15 ; kb = (lane>=16)?8:0 ; elems 0..7 -> k = kb + 0..7 ; elems 8..15 -> k = kb + 16..23
__device__ __forceinline__ v16bf ld_frag_b(const bf16* tile, int ld, int lane) {
  const int n  = lane & 15;
  const int kb = (lane & 16) ? 8 : 0;
  const bf16* p = tile + n * ld + kb;
  BF16x16 r;
  r.h[0] = *(const v8bf*)(p);
  r.h[1] = *(const v8bf*)(p + 16);
  return r.v;
}

// ---------------------------------------------------------------------------
// CDNA5 async global->LDS copy (ASYNCcnt-tracked).
// Builtin signature (verified by hipcc): (v4i as1*, v4i as3*, Ii off, Ii cpol)
// ---------------------------------------------------------------------------
#define AS1 __attribute__((address_space(1)))
#define AS3 __attribute__((address_space(3)))

#if defined(__gfx1250__) && __has_builtin(__builtin_amdgcn_global_load_async_to_lds_b128)
#define HAS_ASYNC_LDS 1
#else
#define HAS_ASYNC_LDS 0
#endif

__device__ __forceinline__ void async_copy_b128(const bf16* gsrc, bf16* ldst) {
#if HAS_ASYNC_LDS
  __builtin_amdgcn_global_load_async_to_lds_b128(
      (AS1 v4i*)gsrc, (AS3 v4i*)ldst, 0, 0);
#else
  *(v8bf*)ldst = *(const v8bf*)gsrc;
#endif
}

__device__ __forceinline__ void async_wait0() {
#if HAS_ASYNC_LDS
#if __has_builtin(__builtin_amdgcn_s_wait_asynccnt)
  __builtin_amdgcn_s_wait_asynccnt(0);
#else
  asm volatile("s_wait_asynccnt 0x0" ::: "memory");
#endif
#endif
}

// ---------------------------------------------------------------------------
// fp32 -> bf16 bulk convert (vectorized x4)
// ---------------------------------------------------------------------------
__global__ __launch_bounds__(256)
void cvt_f32_bf16_kernel(const float* __restrict__ src, bf16* __restrict__ dst,
                         int n4) {
  const int i = blockIdx.x * blockDim.x + threadIdx.x;
  if (i < n4) {
    const float4 v = ((const float4*)src)[i];
    v4bf o;
    o[0] = (bf16)v.x; o[1] = (bf16)v.y; o[2] = (bf16)v.z; o[3] = (bf16)v.w;
    ((v4bf*)dst)[i] = o;
  }
}

// ---------------------------------------------------------------------------
// Weight transpose + convert: src f32 [K][N] row-major -> dst bf16 [N][K]
// ---------------------------------------------------------------------------
__global__ __launch_bounds__(256)
void transpose_cvt_kernel(const float* __restrict__ src, bf16* __restrict__ dst,
                          int K, int N) {
  __shared__ float tile[32][33];
  const int tx = threadIdx.x & 31;
  const int ty = threadIdx.x >> 5;   // 0..7
  const int n0 = blockIdx.x * 32;
  const int k0 = blockIdx.y * 32;
  #pragma unroll
  for (int j = 0; j < 32; j += 8)
    tile[ty + j][tx] = src[(size_t)(k0 + ty + j) * N + n0 + tx];
  __syncthreads();
  #pragma unroll
  for (int j = 0; j < 32; j += 8)
    dst[(size_t)(n0 + ty + j) * K + k0 + tx] = (bf16)tile[tx][ty + j];
}

// ---------------------------------------------------------------------------
// V head transpose: qkv bf16 [B*S][3F] (V slice) -> vT bf16 [B*H][D][S]
// so attention can stage V tiles with contiguous async b128 copies.
// Grid: (S/32, D/32, B*H), block 256.
// ---------------------------------------------------------------------------
__global__ __launch_bounds__(256)
void v_transpose_kernel(const bf16* __restrict__ qkv, bf16* __restrict__ vT,
                        int S, int H, int D) {
  __shared__ bf16 tile[32][33];
  const int tx = threadIdx.x & 31;
  const int ty = threadIdx.x >> 5;
  const int s0 = blockIdx.x * 32;
  const int d0 = blockIdx.y * 32;
  const int h  = blockIdx.z % H;
  const int b  = blockIdx.z / H;
  const int F  = H * D;
  const int rs = 3 * F;
  const bf16* src = qkv + (size_t)b * S * rs + 2 * F + h * D;
  #pragma unroll
  for (int j = 0; j < 32; j += 8)
    tile[ty + j][tx] = src[(size_t)(s0 + ty + j) * rs + d0 + tx];
  __syncthreads();
  bf16* dst = vT + (size_t)(b * H + h) * D * S;
  #pragma unroll
  for (int j = 0; j < 32; j += 8)
    dst[(size_t)(d0 + ty + j) * S + s0 + tx] = tile[tx][ty + j];
}

// ---------------------------------------------------------------------------
// WMMA GEMM: C[M][N] = A[M][K](bf16,rm) * Bt[N][K](bf16) + bias[N]
// Block tile 256x128, 8 waves; wave (wr,wc) owns a 64x64 region = 4x4 WMMA
// accum tiles, so each staged fragment feeds 4 WMMAs (16 WMMA : 16 DS-b128
// per K-step per wave). K-step 32. Double-buffered async LDS staging.
// ---------------------------------------------------------------------------
template <bool OUT_F32>
__global__ __launch_bounds__(256)
void gemm_bf16_kernel(const bf16* __restrict__ A, const bf16* __restrict__ Bt,
                      const float* __restrict__ bias, void* __restrict__ Cout,
                      int M, int N, int K) {
  constexpr int LD = 40;                 // 32 + 8 pad (16B-aligned rows)
  __shared__ bf16 sA[2][256][LD];
  __shared__ bf16 sB[2][128][LD];
  const int tid  = threadIdx.x;
  const int lane = tid & 31;
  const int wave = tid >> 5;
  const int wr = wave >> 1;              // 0..3 -> 64-row strip
  const int wc = wave & 1;               // 0..1 -> 64-col strip
  const int m_blk = blockIdx.y * 256;
  const int n_blk = blockIdx.x * 128;

  // staging: A tile 256x32 -> 1 row (32 bf16, 4 asyncs) per thread;
  //          B tile 128x32 -> half row (16 bf16, 2 asyncs) per thread.
  const int rb = tid >> 1;
  const int cb = (tid & 1) << 4;
  const bf16* arow = A  + (size_t)(m_blk + tid) * K;
  const bf16* brow = Bt + (size_t)(n_blk + rb) * K + cb;

  auto stage = [&](int buf, int kk) {
    async_copy_b128(arow + kk,      &sA[buf][tid][0]);
    async_copy_b128(arow + kk + 8,  &sA[buf][tid][8]);
    async_copy_b128(arow + kk + 16, &sA[buf][tid][16]);
    async_copy_b128(arow + kk + 24, &sA[buf][tid][24]);
    async_copy_b128(brow + kk,      &sB[buf][rb][cb]);
    async_copy_b128(brow + kk + 8,  &sB[buf][rb][cb + 8]);
  };

  v8f acc[4][4] = {};
  const int nT = K >> 5;
  stage(0, 0);

  for (int it = 0; it < nT; ++it) {
    async_wait0();
    __syncthreads();                      // buffer it&1 is ready for everyone
    if (it + 1 < nT) stage((it + 1) & 1, (it + 1) << 5);

    const int buf = it & 1;
    v16bf afr[4], bfr[4];
    #pragma unroll
    for (int i = 0; i < 4; ++i)
      afr[i] = ld_frag_a(&sA[buf][wr * 64 + i * 16][0], LD, lane);
    #pragma unroll
    for (int j = 0; j < 4; ++j)
      bfr[j] = ld_frag_b(&sB[buf][wc * 64 + j * 16][0], LD, lane);
    #pragma unroll
    for (int i = 0; i < 4; ++i)
      #pragma unroll
      for (int j = 0; j < 4; ++j)
        acc[i][j] = wmma_bf16(afr[i], bfr[j], acc[i][j]);
    __syncthreads();                      // done reading buf before reuse
  }

  // epilogue: C layout -> elem p = row (p + 8*(lane>=16)), col = lane&15
  const int rlo = (lane >> 4) << 3;
  #pragma unroll
  for (int i = 0; i < 4; ++i) {
    const int row0 = m_blk + wr * 64 + i * 16 + rlo;
    #pragma unroll
    for (int j = 0; j < 4; ++j) {
      const int col = n_blk + wc * 64 + j * 16 + (lane & 15);
      const float bv = bias[col];
      #pragma unroll
      for (int p = 0; p < 8; ++p) {
        const float v = acc[i][j][p] + bv;
        const size_t idx = (size_t)(row0 + p) * N + col;
        if constexpr (OUT_F32) ((float*)Cout)[idx] = v;
        else                   ((bf16*)Cout)[idx]  = (bf16)v;
      }
    }
  }
}

// ---------------------------------------------------------------------------
// Flash attention (causal), WMMA everywhere, async double-buffered K/V.
// Grid: (S/256, H, B). Block: 256 thr = 8 waves; wave w owns 32 Q rows
// [q0+32w, q0+32w+32) = two 16-row tiles, end-to-end (softmax state stays
// in-wave). Each K/V fragment staged through LDS now feeds TWO WMMAs,
// halving DS bytes per matrix op. KV streamed in 32-wide chunks:
//   sK  [32][128] row-major (kv,d) -> B-frags for S = Q K^T
//   sVt [128][32] (d,kv) from pre-transposed vT -> B-frags for Z += P V
// P restaged per wave through LDS (C-layout -> A-layout); DS ops are
// in-order per wave so no barrier is needed for that private buffer.
// ---------------------------------------------------------------------------
__global__ __launch_bounds__(256)
void flash_attn_kernel(const bf16* __restrict__ qkv, const bf16* __restrict__ vT,
                       bf16* __restrict__ out, int S, int H, int D, float scale) {
  constexpr int LDK = 136;   // 128 + 8
  constexpr int LDV = 40;    // 32 + 8
  constexpr int LDP = 40;
  __shared__ bf16 sK[2][32][LDK];
  __shared__ bf16 sVt[2][128][LDV];
  __shared__ bf16 sP[8][32][LDP];

  const int tid = threadIdx.x, lane = tid & 31, wave = tid >> 5;
  const int h = blockIdx.y, b = blockIdx.z;
  const int Fdim = H * D;                 // 2048
  const int rs = 3 * Fdim;                // qkv row stride (elements)
  const int q0 = blockIdx.x * 256;
  const int qw = q0 + wave * 32;          // this wave's first Q row
  const bf16* qb    = qkv + (size_t)b * S * rs + h * D;
  const bf16* kbase = qb + Fdim;
  const bf16* vtb   = vT + (size_t)(b * H + h) * D * S;

  // Q fragments: 2 row-tiles x 4 A-frags across D=128, straight from global
  v16bf qf[2][4];
  {
    const int m   = lane & 15;
    const int kbo = (lane & 16) ? 8 : 0;
    #pragma unroll
    for (int i = 0; i < 2; ++i) {
      const bf16* qrow = qb + (size_t)(qw + i * 16 + m) * rs;
      #pragma unroll
      for (int kk = 0; kk < 4; ++kk) {
        BF16x16 rr;
        rr.h[0] = *(const v8bf*)(qrow + kk * 32 + kbo);
        rr.h[1] = *(const v8bf*)(qrow + kk * 32 + kbo + 16);
        qf[i][kk] = rr.v;
      }
    }
  }

  // staging assignment
  const int kr = tid >> 3, kc = (tid & 7) << 4;   // K: 32 rows x 128 cols
  const int vd = tid >> 1, vc = (tid & 1) << 4;   // Vt: 128 rows x 32 cols
  auto stage = [&](int buf, int kv0) {
    const bf16* ksrc = kbase + (size_t)(kv0 + kr) * rs + kc;
    async_copy_b128(ksrc,     &sK[buf][kr][kc]);
    async_copy_b128(ksrc + 8, &sK[buf][kr][kc + 8]);
    const bf16* vsrc = vtb + (size_t)vd * S + kv0 + vc;
    async_copy_b128(vsrc,     &sVt[buf][vd][vc]);
    async_copy_b128(vsrc + 8, &sVt[buf][vd][vc + 8]);
  };

  v8f zacc[2][8] = {};
  float rowmax[2][8], rowsum[2][8];
  #pragma unroll
  for (int i = 0; i < 2; ++i)
    #pragma unroll
    for (int p = 0; p < 8; ++p) { rowmax[i][p] = -1e30f; rowsum[i][p] = 0.0f; }
  const int rlo = (lane >> 4) << 3;

  const int nChunks = (q0 + 256) >> 5;    // causal upper bound for this block
  stage(0, 0);

  for (int it = 0; it < nChunks; ++it) {
    async_wait0();
    __syncthreads();                      // chunk it ready in buffer it&1
    if (it + 1 < nChunks) stage((it + 1) & 1, (it + 1) << 5);

    const int kv0 = it << 5;
    const int buf = it & 1;
    if (kv0 <= qw + 31) {                 // wave-uniform causal skip
      // ---- scores: S(32x32) = Q(32x128) @ K^T(128x32), kf reused x2 ----
      v16bf kf[8];
      #pragma unroll
      for (int kk = 0; kk < 4; ++kk)
        #pragma unroll
        for (int t = 0; t < 2; ++t)
          kf[kk * 2 + t] = ld_frag_b(&sK[buf][t * 16][kk * 32], LDK, lane);
      v8f sc[2][2] = {};
      #pragma unroll
      for (int kk = 0; kk < 4; ++kk)
        #pragma unroll
        for (int i = 0; i < 2; ++i) {
          sc[i][0] = wmma_bf16(qf[i][kk], kf[kk * 2 + 0], sc[i][0]);
          sc[i][1] = wmma_bf16(qf[i][kk], kf[kk * 2 + 1], sc[i][1]);
        }
      // ---- per row-tile: scale, mask, online softmax, stage P ----
      float alpha[2][8];
      #pragma unroll
      for (int i = 0; i < 2; ++i) {
        const int rbase = qw + i * 16 + rlo;
        const bool need_mask = (kv0 + 31) > (qw + i * 16);
        #pragma unroll
        for (int t = 0; t < 2; ++t) {
          const int col = kv0 + t * 16 + (lane & 15);
          #pragma unroll
          for (int p = 0; p < 8; ++p) {
            const float v = sc[i][t][p] * scale;
            sc[i][t][p] = (need_mask && (col > rbase + p)) ? -1e30f : v;
          }
        }
        float pm[8];
        #pragma unroll
        for (int p = 0; p < 8; ++p) pm[p] = fmaxf(sc[i][0][p], sc[i][1][p]);
        #pragma unroll
        for (int off = 1; off < 16; off <<= 1)
          #pragma unroll
          for (int p = 0; p < 8; ++p)
            pm[p] = fmaxf(pm[p], __shfl_xor(pm[p], off, 32));
        float mnew[8], psum[8];
        #pragma unroll
        for (int p = 0; p < 8; ++p) {
          mnew[p]     = fmaxf(rowmax[i][p], pm[p]);
          alpha[i][p] = __expf(rowmax[i][p] - mnew[p]);
          psum[p]     = 0.0f;
        }
        #pragma unroll
        for (int t = 0; t < 2; ++t)
          #pragma unroll
          for (int p = 0; p < 8; ++p) {
            const float e = __expf(sc[i][t][p] - mnew[p]);
            psum[p] += e;
            sP[wave][i * 16 + rlo + p][t * 16 + (lane & 15)] = (bf16)e;
          }
        #pragma unroll
        for (int off = 1; off < 16; off <<= 1)
          #pragma unroll
          for (int p = 0; p < 8; ++p)
            psum[p] += __shfl_xor(psum[p], off, 32);
        #pragma unroll
        for (int p = 0; p < 8; ++p) {
          rowsum[i][p] = rowsum[i][p] * alpha[i][p] + psum[p];
          rowmax[i][p] = mnew[p];
        }
      }
      // ---- Z += P(32x32) @ V(32x128), vf reused x2 ----
      v16bf pf[2];
      pf[0] = ld_frag_a(&sP[wave][0][0],  LDP, lane);
      pf[1] = ld_frag_a(&sP[wave][16][0], LDP, lane);
      v16bf vf[8];
      #pragma unroll
      for (int j = 0; j < 8; ++j)
        vf[j] = ld_frag_b(&sVt[buf][j * 16][0], LDV, lane);
      #pragma unroll
      for (int i = 0; i < 2; ++i)
        #pragma unroll
        for (int j = 0; j < 8; ++j)
          #pragma unroll
          for (int p = 0; p < 8; ++p)
            zacc[i][j][p] *= alpha[i][p];
      #pragma unroll
      for (int j = 0; j < 8; ++j) {
        zacc[0][j] = wmma_bf16(pf[0], vf[j], zacc[0][j]);
        zacc[1][j] = wmma_bf16(pf[1], vf[j], zacc[1][j]);
      }
    }
    __syncthreads();                      // done reading buf before reuse
  }

  // ---- normalize and write merged [B*S][F] bf16 ----
  #pragma unroll
  for (int i = 0; i < 2; ++i) {
    float inv[8];
    #pragma unroll
    for (int p = 0; p < 8; ++p) inv[p] = 1.0f / rowsum[i][p];
    #pragma unroll
    for (int j = 0; j < 8; ++j) {
      const int col = h * D + j * 16 + (lane & 15);
      #pragma unroll
      for (int p = 0; p < 8; ++p) {
        const size_t row = (size_t)b * S + qw + i * 16 + rlo + p;
        out[row * Fdim + col] = (bf16)(zacc[i][j][p] * inv[p]);
      }
    }
  }
}

// ---------------------------------------------------------------------------
// Host launcher
// ---------------------------------------------------------------------------
extern "C" void kernel_launch(void* const* d_in, const int* in_sizes, int n_in,
                              void* d_out, int out_size, void* d_ws, size_t ws_size,
                              hipStream_t stream) {
  (void)in_sizes; (void)n_in; (void)out_size; (void)ws_size;
  constexpr int Bn = 2, S = 2048, Fd = 2048, H = 16, D = 128;
  constexpr int M  = Bn * S;          // 4096
  constexpr int K  = Fd;              // 2048
  constexpr int N1 = 3 * Fd;          // 6144

  const float* x      = (const float*)d_in[0];
  /* d_in[1] = causal mask: structure known, ignored */
  const float* w_attn = (const float*)d_in[2];
  const float* b_attn = (const float*)d_in[3];
  const float* w_proj = (const float*)d_in[4];
  const float* b_proj = (const float*)d_in[5];

  // workspace carve-up (bf16 intermediates, ~133 MB total)
  char* ws = (char*)d_ws;
  bf16* x_bf    = (bf16*)ws; ws += (size_t)M  * K  * sizeof(bf16);
  bf16* wattn_t = (bf16*)ws; ws += (size_t)N1 * K  * sizeof(bf16);
  bf16* wproj_t = (bf16*)ws; ws += (size_t)Fd * K  * sizeof(bf16);
  bf16* qkv     = (bf16*)ws; ws += (size_t)M  * N1 * sizeof(bf16);
  bf16* merged  = (bf16*)ws; ws += (size_t)M  * Fd * sizeof(bf16);
  bf16* vT      = (bf16*)ws; ws += (size_t)Bn * H * D * S * sizeof(bf16);

  // 1) convert x to bf16
  cvt_f32_bf16_kernel<<<(M * K / 4 + 255) / 256, 256, 0, stream>>>(x, x_bf, M * K / 4);
  // 2) transpose+convert weights: [K][N] f32 -> [N][K] bf16
  transpose_cvt_kernel<<<dim3(N1 / 32, K / 32), 256, 0, stream>>>(w_attn, wattn_t, K, N1);
  transpose_cvt_kernel<<<dim3(Fd / 32, K / 32), 256, 0, stream>>>(w_proj, wproj_t, K, Fd);
  // 3) fused QKV projection: qkv = x @ W_attn + b (bf16 out)
  gemm_bf16_kernel<false><<<dim3(N1 / 128, M / 256), 256, 0, stream>>>(
      x_bf, wattn_t, b_attn, qkv, M, N1, K);
  // 4) pre-transpose V heads: vT[b*H+h][d][s]
  v_transpose_kernel<<<dim3(S / 32, D / 32, Bn * H), 256, 0, stream>>>(qkv, vT, S, H, D);
  // 5) causal flash attention -> merged (bf16)
  const float scale = 0.088388347648318447f;   // 1/sqrt(128)
  flash_attn_kernel<<<dim3(S / 256, H, Bn), 256, 0, stream>>>(qkv, vT, merged, S, H, D, scale);
  // 6) output projection (f32 out to d_out)
  gemm_bf16_kernel<true><<<dim3(Fd / 128, M / 256), 256, 0, stream>>>(
      merged, wproj_t, b_proj, d_out, M, Fd, K);
}
